// Criterion_BiNet_1211180777885
// MI455X (gfx1250) — compile-verified, matrix operand order checked by reading, not run on hardware
//
#include <hip/hip_runtime.h>
#include <hip/hip_bf16.h>

typedef __attribute__((ext_vector_type(16))) _Float16 v16h;
typedef __attribute__((ext_vector_type(8)))  _Float16 v8h;
typedef __attribute__((ext_vector_type(8)))  float    v8f;
typedef __attribute__((ext_vector_type(4)))  float    v4f;

#define G_RED   4096          // blocks for grid-stride reduction kernels
#define NSLOT   91            // 36 inter + 54 pooled-sums + 1 mono
#define NV      2867200       // 4*28*160*160
#define HW      25600         // 160*160
#define DHW     716800        // 28*160*160

// ---- workspace layout (bytes) ----
#define OFF_BW   0u                       // 6 * 160*160 f16 Gram matrices (H/W axes)
#define OFF_BD   307200u                  // 6 * 28*28 f32 Gram matrices (D axis)
#define OFF_CW   326016u                  // 6 * 160 f32 boundary-count weights (H/W)
#define OFF_CD   329856u                  // 6 * 28  f32 boundary-count weights (D)
#define OFF_ACC  330528u                  // NSLOT f32 final accumulators
#define OFF_PART 331776u                  // NSLOT * G_RED f32 block partials
#define OFF_T1   1822720u                 // NV f32 temp (after W conv)
#define OFF_T2   13291520u                // NV f32 temp (after H conv)
// total ~24.8 MB

__device__ __forceinline__ int iabs(int v) { return v < 0 ? -v : v; }

__device__ __forceinline__ float blockReduce(float v, float* sm) {
  __syncthreads();                 // safe for back-to-back use of same smem
  int tid = threadIdx.x;
  sm[tid] = v;
  __syncthreads();
  for (int s = 128; s > 0; s >>= 1) {
    if (tid < s) sm[tid] += sm[tid + s];
    __syncthreads();
  }
  return sm[0];
}

// ---------------------------------------------------------------------------
// Build per-scale operators on device: B = A^T A Gram matrices per axis
// (triangle kernel tri[d] = k-|d|, pooled = tri/k^2, exact zero-pad boundaries)
// and column-sum weights c_i = sum_j tri[j-i]/k^2.  B is symmetric.
// ---------------------------------------------------------------------------
__global__ __launch_bounds__(256) void build_weights(char* ws) {
  _Float16* Bw = (_Float16*)(ws + OFF_BW);
  float*    Bd = (float*)(ws + OFF_BD);
  float*    cW = (float*)(ws + OFF_CW);
  float*    cD = (float*)(ws + OFF_CD);
  const int KD[6] = {1, 1, 3, 5, 7, 9};
  const int KW[6] = {1, 5, 13, 23, 31, 41};
  int tid = threadIdx.x;
  for (int s = 0; s < 6; ++s) {
    int kw = KW[s], kd = KD[s];
    float k2w = (float)(kw * kw), k4w = k2w * k2w;
    float k2d = (float)(kd * kd), k4d = k2d * k2d;
    for (int idx = tid; idx < 160 * 160; idx += blockDim.x) {
      int i = idx / 160, j = idx % 160;
      float acc = 0.f;
      int lo = max(0, max(i - kw + 1, j - kw + 1));
      int hi = min(159, min(i + kw - 1, j + kw - 1));
      for (int m = lo; m <= hi; ++m)
        acc += (float)((kw - iabs(i - m)) * (kw - iabs(j - m)));
      Bw[s * 25600 + idx] = (_Float16)(acc / k4w);
    }
    for (int idx = tid; idx < 28 * 28; idx += blockDim.x) {
      int i = idx / 28, j = idx % 28;
      float acc = 0.f;
      int lo = max(0, max(i - kd + 1, j - kd + 1));
      int hi = min(27, min(i + kd - 1, j + kd - 1));
      for (int m = lo; m <= hi; ++m)
        acc += (float)((kd - iabs(i - m)) * (kd - iabs(j - m)));
      Bd[s * 784 + idx] = acc / k4d;
    }
    for (int i = tid; i < 160; i += blockDim.x) {
      float acc = 0.f;
      int lo = max(0, i - kw + 1), hi = min(159, i + kw - 1);
      for (int j = lo; j <= hi; ++j) acc += (float)(kw - iabs(j - i));
      cW[s * 160 + i] = acc / k2w;
    }
    for (int i = tid; i < 28; i += blockDim.x) {
      float acc = 0.f;
      int lo = max(0, i - kd + 1), hi = min(27, i + kd - 1);
      for (int j = lo; j <= hi; ++j) acc += (float)(kd - iabs(j - i));
      cD[s * 28 + i] = acc / k2d;
    }
  }
}

// Load 16 f16 (two aligned b128 chunks 0..7 and 16..23 halves apart)
__device__ __forceinline__ v16h load_gram_chunk(const _Float16* p) {
  v8h lo = *(const v8h*)(p);
  v8h hi = *(const v8h*)(p + 16);
  return __builtin_shufflevector(lo, hi, 0, 1, 2, 3, 4, 5, 6, 7,
                                         8, 9, 10, 11, 12, 13, 14, 15);
}

// ---------------------------------------------------------------------------
// W-axis conv as WMMA GEMM: Y[17920 x 160] = X[17920 x 160] * Bw[160 x 160].
// One wave per 16-row strip: volume operand (A) held in registers for all K
// (5 x v16h, loaded as b128 + cvt_pk), Gram matrix streamed as contiguous
// b128 loads using its symmetry (column nb == row nb). 50 WMMAs per wave.
// A layout (16-bit 16x32, ISA 7.12.2): lanes 0-15 row M=lane, k {0..7,16..23};
// lanes 16-31 same rows, k {8..15,24..31}. B mirrored per-column.
// ---------------------------------------------------------------------------
__global__ __launch_bounds__(256) void conv_w_gemm(const float* __restrict__ x,
                                                   const _Float16* __restrict__ Bw,
                                                   float* __restrict__ y) {
  int wave = threadIdx.x >> 5;
  int lane = threadIdx.x & 31;
  int strip = blockIdx.x * 8 + wave;        // grid 140 * 8 waves == 1120 strips
  int m0 = strip * 16;
  int row   = m0 + (lane & 15);
  int khalf = (lane & 16) ? 8 : 0;
  const float* arow = x + row * 160;
  v16h aS[5];
#pragma unroll
  for (int k = 0; k < 5; ++k) {
    int kb = k * 32 + khalf;
    v4f a0 = *(const v4f*)(arow + kb);
    v4f a1 = *(const v4f*)(arow + kb + 4);
    v4f a2 = *(const v4f*)(arow + kb + 16);
    v4f a3 = *(const v4f*)(arow + kb + 20);
#pragma unroll
    for (int i = 0; i < 4; ++i) {
      aS[k][i]      = (_Float16)a0[i];
      aS[k][4 + i]  = (_Float16)a1[i];
      aS[k][8 + i]  = (_Float16)a2[i];
      aS[k][12 + i] = (_Float16)a3[i];
    }
  }
  int mS = m0 + ((lane & 16) ? 8 : 0);
  for (int n = 0; n < 10; ++n) {
    int nb = n * 16 + (lane & 15);
    const _Float16* bcol = Bw + nb * 160;   // symmetric: column == row
    v8f acc = {};
#pragma unroll
    for (int k = 0; k < 5; ++k) {
      v16h b = load_gram_chunk(bcol + k * 32 + khalf);
      acc = __builtin_amdgcn_wmma_f32_16x16x32_f16(false, aS[k], false, b,
                                                   (short)0, acc, false, false);
    }
#pragma unroll
    for (int v = 0; v < 8; ++v) y[(mS + v) * 160 + nb] = acc[v];
  }
}

// H-axis conv: per (b,d) slice, Y = Bw * X. One wave per (slice, 16-w-column
// block): volume operand (B) held in registers for all K, Gram matrix (A)
// streamed as contiguous b128 row loads while looping the 10 h'-tiles.
__global__ __launch_bounds__(256) void conv_h_gemm(const float* __restrict__ x,
                                                   const _Float16* __restrict__ Bw,
                                                   float* __restrict__ y) {
  int wave = threadIdx.x >> 5;
  int lane = threadIdx.x & 31;
  int strip = blockIdx.x * 8 + wave;        // 112 slices * 10 = 1120 strips
  int slice = strip / 10;
  int n0    = (strip % 10) * 16;
  int nb    = n0 + (lane & 15);
  int khalf = (lane & 16) ? 8 : 0;
  const float* xs = x + slice * HW;
  v16h bS[5];
#pragma unroll
  for (int k = 0; k < 5; ++k) {
    int kb = k * 32 + khalf;
#pragma unroll
    for (int i = 0; i < 8; ++i) {
      bS[k][i]     = (_Float16)xs[(kb + i) * 160 + nb];
      bS[k][8 + i] = (_Float16)xs[(kb + 16 + i) * 160 + nb];
    }
  }
  for (int m = 0; m < 10; ++m) {
    const _Float16* arow = Bw + (m * 16 + (lane & 15)) * 160;
    v8f acc = {};
#pragma unroll
    for (int k = 0; k < 5; ++k) {
      v16h a = load_gram_chunk(arow + k * 32 + khalf);
      acc = __builtin_amdgcn_wmma_f32_16x16x32_f16(false, a, false, bS[k],
                                                   (short)0, acc, false, false);
    }
    int hS = m * 16 + ((lane & 16) ? 8 : 0);
#pragma unroll
    for (int v = 0; v < 8; ++v) y[(slice * 160 + hS + v) * 160 + nb] = acc[v];
  }
}

// D-axis conv (28x28 Gram from LDS) fused with both inter-dot-products.
// One thread per (b,h,w) column: the 28-deep column is loaded once
// (lane-coalesced per d), then all 28 outputs produced in registers.
__global__ __launch_bounds__(256) void dconv_dot(const float* __restrict__ t2,
                                                 const float* __restrict__ pc,
                                                 const float* __restrict__ pp,
                                                 const float* __restrict__ Bd,
                                                 float* __restrict__ part,
                                                 int slot1, int slot2) {
  __shared__ float sBd[784];
  for (int i = threadIdx.x; i < 784; i += blockDim.x) sBd[i] = Bd[i];
  __syncthreads();
  const int NC = 4 * HW;                    // 102400 columns
  float s1 = 0.f, s2 = 0.f;
  for (int idx = blockIdx.x * blockDim.x + threadIdx.x; idx < NC;
       idx += gridDim.x * blockDim.x) {
    int b  = idx / HW;
    int hw = idx % HW;
    const float* base = t2 + b * DHW + hw;
    float col[28];
#pragma unroll
    for (int d = 0; d < 28; ++d) col[d] = base[d * HW];
    const float* pcb = pc + b * DHW + hw;
    const float* ppb = pp + b * DHW + hw;
    for (int dp = 0; dp < 28; ++dp) {
      const float* br = sBd + dp * 28;
      float q = 0.f;
#pragma unroll
      for (int d = 0; d < 28; ++d) q += br[d] * col[d];
      s1 += q * pcb[dp * HW];
      s2 += q * ppb[dp * HW];
    }
  }
  __shared__ float sm[256];
  float r1 = blockReduce(s1, sm);
  float r2 = blockReduce(s2, sm);
  if (threadIdx.x == 0) {
    part[slot1 * G_RED + blockIdx.x] = r1;
    part[slot2 * G_RED + blockIdx.x] = r2;
  }
}

// Pooled-field sums of all 9 volumes for all 6 scales in one pass
// (separable boundary weights staged in LDS). Slots 36 + s*9 + t.
__global__ __launch_bounds__(256) void wsum_all(const float* __restrict__ t0, const float* __restrict__ t1,
                                                const float* __restrict__ t2, const float* __restrict__ t3,
                                                const float* __restrict__ t4, const float* __restrict__ t5,
                                                const float* __restrict__ t6, const float* __restrict__ t7,
                                                const float* __restrict__ t8,
                                                const float* __restrict__ cW, const float* __restrict__ cD,
                                                float* __restrict__ part) {
  __shared__ float sW[6 * 160];
  __shared__ float sD[6 * 28];
  for (int i = threadIdx.x; i < 6 * 160; i += blockDim.x) sW[i] = cW[i];
  for (int i = threadIdx.x; i < 6 * 28; i += blockDim.x) sD[i] = cD[i];
  __syncthreads();
  const float* T[9] = {t0, t1, t2, t3, t4, t5, t6, t7, t8};
  float acc[54];
#pragma unroll
  for (int k = 0; k < 54; ++k) acc[k] = 0.f;
  for (int idx = blockIdx.x * blockDim.x + threadIdx.x; idx < NV;
       idx += gridDim.x * blockDim.x) {
    int r  = idx % DHW;
    int d  = r / HW;
    int hw = r % HW;
    int h  = hw / 160, w = hw % 160;
    float xv[9];
#pragma unroll
    for (int t = 0; t < 9; ++t) xv[t] = T[t][idx];
#pragma unroll
    for (int s = 0; s < 6; ++s) {
      float wgt = sD[s * 28 + d] * sW[s * 160 + h] * sW[s * 160 + w];
#pragma unroll
      for (int t = 0; t < 9; ++t) acc[s * 9 + t] += xv[t] * wgt;
    }
  }
  __shared__ float sm[256];
  for (int k = 0; k < 54; ++k) {
    float r = blockReduce(acc[k], sm);
    if (threadIdx.x == 0) part[(36 + k) * G_RED + blockIdx.x] = r;
  }
}

// Temporal monotonicity: sum over all (|diff| - diff), diff across channel axis.
__global__ __launch_bounds__(256) void mono_kernel(const float* __restrict__ o,
                                                   float* __restrict__ part, int slot) {
  const int TOT = 4 * 5 * DHW;
  float s = 0.f;
  for (int idx = blockIdx.x * blockDim.x + threadIdx.x; idx < TOT;
       idx += gridDim.x * blockDim.x) {
    int b = idx / (5 * DHW);
    int r = idx % (5 * DHW);
    int t = r / DHW;
    int v = r % DHW;
    float d = o[(b * 6 + t + 1) * DHW + v] - o[(b * 6 + t) * DHW + v];
    s += fabsf(d) - d;
  }
  __shared__ float sm[256];
  float rr = blockReduce(s, sm);
  if (threadIdx.x == 0) part[slot * G_RED + blockIdx.x] = rr;
}

// Deterministic second-stage reduction: one block per slot, fixed order.
__global__ __launch_bounds__(256) void reduce_slots(const float* __restrict__ part,
                                                    float* __restrict__ acc) {
  int slot = blockIdx.x;
  float s = 0.f;
  for (int i = threadIdx.x; i < G_RED; i += 256) s += part[slot * G_RED + i];
  __shared__ float sm[256];
  float r = blockReduce(s, sm);
  if (threadIdx.x == 0) acc[slot] = r;
}

__global__ void finalize(const float* __restrict__ acc,
                         const float* __restrict__ off_core_c,
                         const float* __restrict__ off_penu_p,
                         const float* __restrict__ off_target_c,
                         const float* __restrict__ off_target_p,
                         float* __restrict__ out) {
  if (threadIdx.x != 0 || blockIdx.x != 0) return;
  const int P[6]  = {0, 1, 3, 4, 6, 7};
  const int GT[6] = {2, 2, 5, 5, 8, 8};
  float loss = 0.f;
  for (int j = 0; j < 6; ++j) {
    float ms = 0.f;
    for (int s = 0; s < 6; ++s) {
      float inter = acc[s * 6 + j];
      float sp = acc[36 + s * 9 + P[j]];
      float st = acc[36 + s * 9 + GT[j]];
      ms += 1.f - (2.f * inter) / (sp + st + 1e-7f);
    }
    loss += 0.2f * (ms / 6.f);
  }
  loss += 0.1f * acc[90] / (float)NV;
  float l1a = 0.f, l1b = 0.f;
  for (int i = 0; i < 12; ++i) {
    l1a += fabsf(off_core_c[i] - off_target_c[i]);
    l1b += fabsf(off_penu_p[i] - off_target_p[i]);
  }
  loss += 0.1f * (l1a / 12.f) + 0.1f * (l1b / 12.f);
  out[0] = loss;
}

extern "C" void kernel_launch(void* const* d_in, const int* in_sizes, int n_in,
                              void* d_out, int out_size, void* d_ws, size_t ws_size,
                              hipStream_t stream) {
  (void)in_sizes; (void)n_in; (void)out_size; (void)ws_size;
  const float* in[10];
  for (int i = 0; i < 10; ++i) in[i] = (const float*)d_in[i];
  char* ws = (char*)d_ws;
  _Float16* Bw  = (_Float16*)(ws + OFF_BW);
  float*    Bd  = (float*)(ws + OFF_BD);
  float*    cWp = (float*)(ws + OFF_CW);
  float*    cDp = (float*)(ws + OFF_CD);
  float*    ACC = (float*)(ws + OFF_ACC);
  float*    PART = (float*)(ws + OFF_PART);
  float*    tmp1 = (float*)(ws + OFF_T1);
  float*    tmp2 = (float*)(ws + OFF_T2);

  build_weights<<<1, 256, 0, stream>>>(ws);

  for (int s = 0; s < 6; ++s) {
    for (int g = 0; g < 3; ++g) {
      const float* gt = in[3 * g + 2];
      conv_w_gemm<<<140, 256, 0, stream>>>(gt, Bw + s * 25600, tmp1);
      conv_h_gemm<<<140, 256, 0, stream>>>(tmp1, Bw + s * 25600, tmp2);
      dconv_dot<<<G_RED, 256, 0, stream>>>(tmp2, in[3 * g], in[3 * g + 1],
                                           Bd + s * 784, PART,
                                           s * 6 + 2 * g, s * 6 + 2 * g + 1);
    }
  }
  wsum_all<<<G_RED, 256, 0, stream>>>(in[0], in[1], in[2], in[3], in[4],
                                      in[5], in[6], in[7], in[8],
                                      cWp, cDp, PART);
  mono_kernel<<<G_RED, 256, 0, stream>>>(in[9], PART, 90);
  reduce_slots<<<NSLOT, 256, 0, stream>>>(PART, ACC);
  finalize<<<1, 64, 0, stream>>>(ACC, (const float*)d_in[10], (const float*)d_in[11],
                                 (const float*)d_in[12], (const float*)d_in[13],
                                 (float*)d_out);
}
